// Encoder_43215960932923
// MI455X (gfx1250) — matrix operand order repaired and checked by reading, not compiled, not run on hardware
//
#include <hip/hip_runtime.h>

// ---------------------------------------------------------------------------
// Types / helpers for CDNA5 (gfx1250) wave32 WMMA
// ---------------------------------------------------------------------------
typedef unsigned short u16;
typedef __attribute__((ext_vector_type(16))) __bf16 v16bf;
typedef __attribute__((ext_vector_type(8)))  float  v8f;
typedef __attribute__((ext_vector_type(4)))  unsigned int u32x4;
typedef __attribute__((ext_vector_type(8)))  unsigned int u32x8;
typedef __attribute__((ext_vector_type(2)))  int i32x2;
typedef __attribute__((ext_vector_type(4)))  int i32x4;

#define AS1 __attribute__((address_space(1)))
#define AS3 __attribute__((address_space(3)))

// gfx1250 async load-to-LDS path (ASYNCcnt), guarded so fallback still compiles
#if defined(__has_builtin)
#if __has_builtin(__builtin_amdgcn_global_load_async_to_lds_b64) && \
    __has_builtin(__builtin_amdgcn_global_load_async_to_lds_b128)
#define HAVE_ASYNC_LDS 1
#endif
#endif
#ifndef HAVE_ASYNC_LDS
#define HAVE_ASYNC_LDS 0
#endif

#if HAVE_ASYNC_LDS
__device__ inline void wait_async0() {
#if __has_builtin(__builtin_amdgcn_s_wait_asynccnt)
  __builtin_amdgcn_s_wait_asynccnt(0);
#else
  asm volatile("s_wait_asynccnt 0x0" ::: "memory");
#endif
}

__device__ inline void async_copy_b64(const u16* gsrc, u16* ldst) {
  __builtin_amdgcn_global_load_async_to_lds_b64((AS1 i32x2*)gsrc,
                                                (AS3 i32x2*)ldst, 0, 0);
}

__device__ inline void async_copy_b128(const u16* gsrc, u16* ldst) {
  __builtin_amdgcn_global_load_async_to_lds_b128((AS1 i32x4*)gsrc,
                                                 (AS3 i32x4*)ldst, 0, 0);
}
#endif

__device__ inline u16 f2bf(float f) {
  unsigned u = __builtin_bit_cast(unsigned, f);
  return (u16)((u + 0x7FFFu + ((u >> 16) & 1u)) >> 16);  // round-to-nearest-even
}

__device__ inline v8f wmma_bf16(v16bf a, v16bf b, v8f c) {
  // D(16x16 f32) = A(16x32 bf16) * B(32x16 bf16) + C
  return __builtin_amdgcn_wmma_f32_16x16x32_bf16(false, a, false, b, (short)0, c,
                                                 false, false);
}

// Load one lane's 16 contiguous bf16 operand values (32B, 16B-aligned) from LDS.
// Lowers to two ds_load_b128.
__device__ inline v16bf load_frag16(const u16* p) {
  u32x4 lo = *(const u32x4*)p;
  u32x4 hi = *(const u32x4*)(p + 8);
  u32x8 w;
#pragma unroll
  for (int i = 0; i < 4; ++i) { w[i] = lo[i]; w[i + 4] = hi[i]; }
  return __builtin_bit_cast(v16bf, w);
}

// Fragment-major swizzle maps (per CDNA5 ISA §7.12.2, wave32):
//  A (16x32, MxK): lane = h*16 + m with h = (k>>3)&1 ; element e = (k>>4)*8 + (k&7)
//  B (32x16, KxN): lane = h*16 + n with h = (k>>3)&1 ; element e = (k>>4)*8 + (k&7)

// ---------------------------------------------------------------------------
// Embedding + sinusoidal positional encoding -> f32 and bf16 activations
// ---------------------------------------------------------------------------
__global__ void embed_pe_kernel(const int* __restrict__ tokens,
                                const float* __restrict__ emb,
                                float* __restrict__ xf, u16* __restrict__ xb,
                                int S, int D, int total) {
  int idx = blockIdx.x * blockDim.x + threadIdx.x;
  if (idx >= total) return;
  int d = idx % D;
  int t = idx / D;      // flat (b,s)
  int s = t % S;
  float v = emb[(size_t)tokens[t] * D + d];
  float i2 = (float)((d >> 1) << 1);
  float angle = (float)s * __powf(10000.0f, -i2 / (float)D);
  v += (d & 1) ? __cosf(angle) : __sinf(angle);
  xf[idx] = v;
  xb[idx] = f2bf(v);
}

// ---------------------------------------------------------------------------
// WMMA GEMM:  C[M,N] = A[M,K](bf16) @ W[K,N](f32 -> bf16 in LDS) + bias
// Block: 256 threads = 8 waves; tile 128x64, K-step 32. Wave -> 32x32 (2x2 frags).
// LDS tiles stored fragment-major: As[mtile][lane][e], Bs[ntile][lane][e].
// A tile staged via GLOBAL_LOAD_ASYNC_TO_LDS_B64 (per-lane swizzled dst).
// ---------------------------------------------------------------------------
__global__ __launch_bounds__(256) void gemm_bias_kernel(
    const u16* __restrict__ A, const float* __restrict__ W,
    const float* __restrict__ bias, float* __restrict__ Cf,
    u16* __restrict__ Cb, int N, int K, int relu) {
  __shared__ __align__(32) u16 As[8][32][16];  // 8 mtiles of 16 rows
  __shared__ __align__(32) u16 Bs[4][32][16];  // 4 ntiles of 16 cols
  int tid = threadIdx.x;
  int wave = tid >> 5, lane = tid & 31;
  int h16 = lane >> 4, ln = lane & 15;
  int wm = wave >> 1, wn = wave & 1;
  int bm0 = blockIdx.y * 128, bn0 = blockIdx.x * 64;

  // Staging indices (A): each thread moves 4 rows x 4 bf16
  int arow = tid >> 3;           // 0..31 (+i*32)
  int acol = (tid & 7) * 4;      // k within 32-panel, multiple of 4
  int a_h = (acol >> 3) & 1;
  int a_e = ((acol >> 4) << 3) + (acol & 7);  // element base (mult of 4)
  // Staging indices (W): each thread moves 2 rows x 4 f32
  int brow = tid >> 4;           // k, 0..15 (+i*16)
  int bcol = (tid & 15) * 4;     // n within 64, multiple of 4

  v8f c[2][2] = {};

  for (int k0 = 0; k0 < K; k0 += 32) {
    __syncthreads();
    if (k0 + 32 < K) {  // prefetch next K-panel (global_prefetch_b8)
      __builtin_prefetch(A + (size_t)(bm0 + arow) * K + k0 + 32 + acol, 0, 1);
      __builtin_prefetch(W + (size_t)(k0 + 32 + brow) * N + bn0 + bcol, 0, 1);
    }
    // Stage A (bf16) 128x32, swizzled: 4-element (8B) groups stay contiguous
#pragma unroll
    for (int i = 0; i < 4; ++i) {
      int row = arow + i * 32;
      const u16* gsrc = A + (size_t)(bm0 + row) * K + k0 + acol;
      u16* ldst = &As[row >> 4][a_h * 16 + (row & 15)][a_e];
#if HAVE_ASYNC_LDS
      async_copy_b64(gsrc, ldst);
#else
      *(uint2*)ldst = *(const uint2*)gsrc;
#endif
    }
    // Stage W (f32 -> bf16) 32x64, swizzled (scatter across 4 lanes, same e)
#pragma unroll
    for (int i = 0; i < 2; ++i) {
      int row = brow + i * 16;                 // k
      int bh = (row >> 3) & 1, be = ((row >> 4) << 3) + (row & 7);
      float4 wv = *(const float4*)(W + (size_t)(k0 + row) * N + bn0 + bcol);
      int nt = bcol >> 4, n0 = bcol & 15;
      Bs[nt][bh * 16 + n0 + 0][be] = f2bf(wv.x);
      Bs[nt][bh * 16 + n0 + 1][be] = f2bf(wv.y);
      Bs[nt][bh * 16 + n0 + 2][be] = f2bf(wv.z);
      Bs[nt][bh * 16 + n0 + 3][be] = f2bf(wv.w);
    }
#if HAVE_ASYNC_LDS
    wait_async0();
#endif
    __syncthreads();

    v16bf a0 = load_frag16(&As[wm * 2 + 0][lane][0]);
    v16bf a1 = load_frag16(&As[wm * 2 + 1][lane][0]);
    v16bf b0 = load_frag16(&Bs[wn * 2 + 0][lane][0]);
    v16bf b1 = load_frag16(&Bs[wn * 2 + 1][lane][0]);
    c[0][0] = wmma_bf16(a0, b0, c[0][0]);
    c[0][1] = wmma_bf16(a0, b1, c[0][1]);
    c[1][0] = wmma_bf16(a1, b0, c[1][0]);
    c[1][1] = wmma_bf16(a1, b1, c[1][1]);
  }

  // Epilogue: bias (+ReLU) applied in-register, then branch-hoisted stores
#pragma unroll
  for (int ni = 0; ni < 2; ++ni) {
    float bv = bias[bn0 + wn * 32 + ni * 16 + ln];
#pragma unroll
    for (int mi = 0; mi < 2; ++mi)
#pragma unroll
      for (int r = 0; r < 8; ++r) c[mi][ni][r] += bv;
  }
  if (relu) {
#pragma unroll
    for (int mi = 0; mi < 2; ++mi)
#pragma unroll
      for (int ni = 0; ni < 2; ++ni)
#pragma unroll
        for (int r = 0; r < 8; ++r) c[mi][ni][r] = fmaxf(c[mi][ni][r], 0.0f);
  }
  int gm0 = bm0 + wm * 32 + 8 * h16;
  int gn0 = bn0 + wn * 32 + ln;
  if (Cf) {
#pragma unroll
    for (int mi = 0; mi < 2; ++mi)
#pragma unroll
      for (int ni = 0; ni < 2; ++ni)
#pragma unroll
        for (int r = 0; r < 8; ++r)
          Cf[(size_t)(gm0 + mi * 16 + r) * N + gn0 + ni * 16] = c[mi][ni][r];
  }
  if (Cb) {
#pragma unroll
    for (int mi = 0; mi < 2; ++mi)
#pragma unroll
      for (int ni = 0; ni < 2; ++ni)
#pragma unroll
        for (int r = 0; r < 8; ++r)
          Cb[(size_t)(gm0 + mi * 16 + r) * N + gn0 + ni * 16] =
              f2bf(c[mi][ni][r]);
  }
}

// ---------------------------------------------------------------------------
// Flash attention, WMMA-based. Grid: (S/64, H, B), 128 threads = 4 waves.
// Each wave owns 16 query rows; loops over key blocks of 32 with online softmax.
// q,k,v,o are bf16 [B,S,D_MODEL] with head h at columns h*64..h*64+63.
// K tile staged via GLOBAL_LOAD_ASYNC_TO_LDS_B128.
// ---------------------------------------------------------------------------
__global__ __launch_bounds__(128) void flash_attn_kernel(
    const u16* __restrict__ q, const u16* __restrict__ k,
    const u16* __restrict__ v, u16* __restrict__ o, int S, int Dm) {
  // Kt row-major with 16B-multiple stride: per-lane B^T fragment reads are
  // 32 contiguous bytes. Vs / Ps are fragment-major swizzled.
  __shared__ __align__(32) u16 Kt[32][72];
  __shared__ __align__(32) u16 Vs[4][32][16];   // [depth tile][lane][e]
  __shared__ __align__(32) u16 Ps[4][32][16];   // [wave][lane][e]

  int tid = threadIdx.x;
  int wave = tid >> 5, lane = tid & 31;
  int h16 = lane >> 4, ln = lane & 15;
  int head = blockIdx.y, batch = blockIdx.z;
  int qrow0 = blockIdx.x * 64 + wave * 16;

  // Q fragments straight from global (one-time), contiguous k = 16*h16 + s (matches Kt B frags)
  const u16* qbase = q + ((size_t)batch * S + qrow0) * Dm + head * 64;
  v16bf qa0, qa1;
  {
    const u16* row = qbase + ln * Dm;
    u32x8 w0, w1;
#pragma unroll
    for (int e = 0; e < 8; ++e) {
      int kk = h16 * 16 + (e << 1);  // k for element 2e
      w0[e] = *(const unsigned int*)(row + kk);
      w1[e] = *(const unsigned int*)(row + 32 + kk);
    }
    qa0 = __builtin_bit_cast(v16bf, w0);
    qa1 = __builtin_bit_cast(v16bf, w1);
  }

  v8f oacc[4] = {};
  float mrow[8], lrow[8];
#pragma unroll
  for (int r = 0; r < 8; ++r) { mrow[r] = -1e30f; lrow[r] = 0.0f; }
  const float scale = 0.125f;  // 1/sqrt(64)

  for (int kb = 0; kb < S / 32; ++kb) {
    int k0 = kb * 32;
    __syncthreads();
#if HAVE_ASYNC_LDS
    // Async copy K block (32 rows x 128B) straight into row-major Kt
#pragma unroll
    for (int i = 0; i < 2; ++i) {
      int cidx = tid + i * 128;          // 0..255 16B-chunks
      int rr = cidx >> 3, j = (cidx & 7) * 8;
      const u16* gsrc = k + ((size_t)batch * S + k0 + rr) * Dm + head * 64 + j;
      async_copy_b128(gsrc, &Kt[rr][j]);
    }
#endif
    // Stage V (swizzled) and, without async support, K too
#pragma unroll
    for (int i = 0; i < 8; ++i) {
      int slot = tid + i * 128;          // 0..1023
      int rr = slot >> 5;                // key row 0..31
      int cc = (slot & 31) * 2;          // depth pair
      size_t src = ((size_t)batch * S + k0 + rr) * Dm + head * 64 + cc;
      unsigned int vv = *(const unsigned int*)(v + src);
#if !HAVE_ASYNC_LDS
      *(unsigned int*)&Kt[rr][cc] = *(const unsigned int*)(k + src);
#endif
      int vh = (rr >> 3) & 1, ve = ((rr >> 4) << 3) + (rr & 7), dt = cc >> 4, n0 = cc & 15;
      Vs[dt][vh * 16 + n0 + 0][ve] = (u16)(vv & 0xFFFFu);
      Vs[dt][vh * 16 + n0 + 1][ve] = (u16)(vv >> 16);
      if (k0 + 32 < S)
        __builtin_prefetch(v + src + (size_t)32 * Dm, 0, 1);
    }
#if HAVE_ASYNC_LDS
    wait_async0();
#endif
    __syncthreads();

    // Scores: S(16x32) = Q(16x64) @ K^T(64x32); B^T frag = contiguous Kt row
    v8f s0 = {}, s1 = {};
    s0 = wmma_bf16(qa0, load_frag16(&Kt[ln][h16 * 16]), s0);
    s0 = wmma_bf16(qa1, load_frag16(&Kt[ln][32 + h16 * 16]), s0);
    s1 = wmma_bf16(qa0, load_frag16(&Kt[16 + ln][h16 * 16]), s1);
    s1 = wmma_bf16(qa1, load_frag16(&Kt[16 + ln][32 + h16 * 16]), s1);

    // Online softmax per row (rows r+8*h16; 16 lanes of a half share a row)
    int p_lane = ((ln >> 3) & 1) * 16;   // A-layout lane group for column ln
    int p_e = ln & 7;
#pragma unroll
    for (int r = 0; r < 8; ++r) {
      float a0 = s0[r] * scale, a1 = s1[r] * scale;
      float mx = fmaxf(a0, a1);
#pragma unroll
      for (int msk = 1; msk < 16; msk <<= 1) mx = fmaxf(mx, __shfl_xor(mx, msk, 32));
      float mnew = fmaxf(mrow[r], mx);
      float alpha = __expf(mrow[r] - mnew);
      float p0 = __expf(a0 - mnew), p1 = __expf(a1 - mnew);
      float ps = p0 + p1;
#pragma unroll
      for (int msk = 1; msk < 16; msk <<= 1) ps += __shfl_xor(ps, msk, 32);
      lrow[r] = lrow[r] * alpha + ps;
      mrow[r] = mnew;
      oacc[0][r] *= alpha; oacc[1][r] *= alpha;
      oacc[2][r] *= alpha; oacc[3][r] *= alpha;
      int m = r + 8 * h16;
      Ps[wave][p_lane + m][p_e]     = f2bf(p0);  // col = ln
      Ps[wave][p_lane + m][p_e + 8] = f2bf(p1);  // col = 16+ln
    }
    __syncthreads();  // make P visible for A-layout reload (uniform across waves)

    // O(16x64) += P(16x32) @ V(32x64)
    v16bf pa = load_frag16(&Ps[wave][lane][0]);
#pragma unroll
    for (int dt = 0; dt < 4; ++dt)
      oacc[dt] = wmma_bf16(pa, load_frag16(&Vs[dt][lane][0]), oacc[dt]);
  }

  // Normalize and store bf16 output
#pragma unroll
  for (int dt = 0; dt < 4; ++dt)
#pragma unroll
    for (int r = 0; r < 8; ++r) {
      float val = oacc[dt][r] / lrow[r];
      int grow = qrow0 + r + 8 * h16;
      int gcol = head * 64 + dt * 16 + ln;
      o[((size_t)batch * S + grow) * Dm + gcol] = f2bf(val);
    }
}

// ---------------------------------------------------------------------------
// out = LayerNorm(x + y) * g + b ; writes f32 and bf16. One row (D=1024)/block.
// ---------------------------------------------------------------------------
__global__ __launch_bounds__(256) void add_ln_kernel(
    const float* __restrict__ x, const float* __restrict__ y,
    const float* __restrict__ g, const float* __restrict__ be,
    float* __restrict__ outf, u16* __restrict__ outb, int D) {
  int row = blockIdx.x;
  const float* xr = x + (size_t)row * D;
  const float* yr = y + (size_t)row * D;
  float vals[4];
  float s = 0.f, s2 = 0.f;
#pragma unroll
  for (int j = 0; j < 4; ++j) {
    int i = threadIdx.x + j * 256;
    float vv = xr[i] + yr[i];
    vals[j] = vv; s += vv; s2 += vv * vv;
  }
#pragma unroll
  for (int msk = 1; msk < 32; msk <<= 1) {
    s += __shfl_xor(s, msk, 32);
    s2 += __shfl_xor(s2, msk, 32);
  }
  __shared__ float ss[8], ssq[8];
  int wave = threadIdx.x >> 5, lane = threadIdx.x & 31;
  if (lane == 0) { ss[wave] = s; ssq[wave] = s2; }
  __syncthreads();
  float ts = 0.f, ts2 = 0.f;
#pragma unroll
  for (int w = 0; w < 8; ++w) { ts += ss[w]; ts2 += ssq[w]; }
  float mean = ts / (float)D;
  float var = ts2 / (float)D - mean * mean;
  float rstd = rsqrtf(var + 1e-6f);
#pragma unroll
  for (int j = 0; j < 4; ++j) {
    int i = threadIdx.x + j * 256;
    float ov = (vals[j] - mean) * rstd * g[i] + be[i];
    outf[(size_t)row * D + i] = ov;
    outb[(size_t)row * D + i] = f2bf(ov);
  }
}

// ---------------------------------------------------------------------------
// Host orchestration
// ---------------------------------------------------------------------------
static void launch_gemm(const u16* A, const float* W, const float* bias,
                        float* Cf, u16* Cb, int M, int N, int K, int relu,
                        hipStream_t st) {
  dim3 grid(N / 64, M / 128);
  gemm_bias_kernel<<<grid, 256, 0, st>>>(A, W, bias, Cf, Cb, N, K, relu);
}

extern "C" void kernel_launch(void* const* d_in, const int* in_sizes, int n_in,
                              void* d_out, int out_size, void* d_ws,
                              size_t ws_size, hipStream_t stream) {
  (void)in_sizes; (void)n_in; (void)out_size; (void)ws_size;

  const int*   tokens = (const int*)d_in[0];
  const float* emb = (const float*)d_in[1];
  const float* Wq = (const float*)d_in[2];  const float* bq = (const float*)d_in[3];
  const float* Wk = (const float*)d_in[4];  const float* bk = (const float*)d_in[5];
  const float* Wv = (const float*)d_in[6];  const float* bv = (const float*)d_in[7];
  const float* Wo = (const float*)d_in[8];  const float* bo = (const float*)d_in[9];
  const float* W1 = (const float*)d_in[10]; const float* b1 = (const float*)d_in[11];
  const float* W2 = (const float*)d_in[12]; const float* b2 = (const float*)d_in[13];
  const float* g1 = (const float*)d_in[14]; const float* be1 = (const float*)d_in[15];
  const float* g2 = (const float*)d_in[16]; const float* be2 = (const float*)d_in[17];

  const int B = 4, S = 1024, D = 1024, H = 16, FF = 4096, L = 6;
  const int M = B * S;  // 4096 tokens

  char* ws = (char*)d_ws;
  size_t off = 0;
  auto alloc = [&](size_t bytes) -> void* {
    void* p = ws + off;
    off += (bytes + 255) & ~(size_t)255;
    return p;
  };
  float* xf  = (float*)alloc((size_t)M * D * 4);   // current activations f32
  u16*   xb  = (u16*)  alloc((size_t)M * D * 2);   // current activations bf16
  u16*   qb  = (u16*)  alloc((size_t)M * D * 2);
  u16*   kb  = (u16*)  alloc((size_t)M * D * 2);
  u16*   vb  = (u16*)  alloc((size_t)M * D * 2);
  u16*   ob  = (u16*)  alloc((size_t)M * D * 2);   // attention output bf16
  float* yf  = (float*)alloc((size_t)M * D * 4);   // projection / ffn result f32
  float* x1f = (float*)alloc((size_t)M * D * 4);   // out1 f32
  u16*   x1b = (u16*)  alloc((size_t)M * D * 2);   // out1 bf16
  u16*   hb  = (u16*)  alloc((size_t)M * FF * 2);  // ffn hidden bf16

  // Embedding + positional encoding
  {
    int total = M * D;
    embed_pe_kernel<<<(total + 255) / 256, 256, 0, stream>>>(tokens, emb, xf, xb,
                                                             S, D, total);
  }

  for (int l = 0; l < L; ++l) {
    const float* Wq_l = Wq + (size_t)l * D * D;  const float* bq_l = bq + (size_t)l * D;
    const float* Wk_l = Wk + (size_t)l * D * D;  const float* bk_l = bk + (size_t)l * D;
    const float* Wv_l = Wv + (size_t)l * D * D;  const float* bv_l = bv + (size_t)l * D;
    const float* Wo_l = Wo + (size_t)l * D * D;  const float* bo_l = bo + (size_t)l * D;
    const float* W1_l = W1 + (size_t)l * D * FF; const float* b1_l = b1 + (size_t)l * FF;
    const float* W2_l = W2 + (size_t)l * FF * D; const float* b2_l = b2 + (size_t)l * D;
    const float* g1_l = g1 + (size_t)l * D;  const float* be1_l = be1 + (size_t)l * D;
    const float* g2_l = g2 + (size_t)l * D;  const float* be2_l = be2 + (size_t)l * D;

    // QKV projections (bf16 outputs only)
    launch_gemm(xb, Wq_l, bq_l, nullptr, qb, M, D, D, 0, stream);
    launch_gemm(xb, Wk_l, bk_l, nullptr, kb, M, D, D, 0, stream);
    launch_gemm(xb, Wv_l, bv_l, nullptr, vb, M, D, D, 0, stream);

    // Attention
    {
      dim3 grid(S / 64, H, B);
      flash_attn_kernel<<<grid, 128, 0, stream>>>(qb, kb, vb, ob, S, D);
    }

    // Output projection (f32) + residual LayerNorm 1
    launch_gemm(ob, Wo_l, bo_l, yf, nullptr, M, D, D, 0, stream);
    add_ln_kernel<<<M, 256, 0, stream>>>(xf, yf, g1_l, be1_l, x1f, x1b, D);

    // FFN: relu(out1 @ W1 + b1) @ W2 + b2
    launch_gemm(x1b, W1_l, b1_l, nullptr, hb, M, FF, D, 1, stream);
    launch_gemm(hb, W2_l, b2_l, yf, nullptr, M, D, FF, 0, stream);

    // Residual LayerNorm 2 -> next layer input (last layer -> d_out)
    float* outf = (l == L - 1) ? (float*)d_out : xf;
    add_ln_kernel<<<M, 256, 0, stream>>>(x1f, yf, g2_l, be2_l, outf, xb, D);
  }
}
